// DynamicSegmentationHead_71665824301696
// MI455X (gfx1250) — matrix operand order repaired and checked by reading, not compiled, not run on hardware
//
#include <hip/hip_runtime.h>
#include <hip/hip_bf16.h>

typedef __attribute__((ext_vector_type(16))) _Float16 v16h;
typedef __attribute__((ext_vector_type(8)))  float    v8f;

#define HID    256
#define NP     169
#define NINST  800
#define QQN    50
#define TTN    8
#define HHN    72
#define WWN    120
#define HWN    (HHN*WWN)      // 8640 pixels / instance
#define NTILES (HWN/16)       // 540 16-pixel tiles / instance
#define NPAIRS (NTILES/2)     // 270 32-pixel tile pairs

static __device__ __forceinline__ v16h hzero16() {
  v16h v;
#pragma unroll
  for (int i = 0; i < 16; ++i) v[i] = (_Float16)0.0f;
  return v;
}
static __device__ __forceinline__ v8f fzero8() {
  v8f v;
#pragma unroll
  for (int i = 0; i < 8; ++i) v[i] = 0.0f;
  return v;
}

union HU {
  v16h h;
  unsigned int u[8];
};

// ---------------------------------------------------------------------------
// Kernel 1: params[800][169] = hs[800][256] @ W_ctrl[169][256]^T + b_ctrl
// One wave per 16x16 output tile; 8 chained v_wmma_f32_16x16x32_f16 over K=256.
// ---------------------------------------------------------------------------
__global__ __launch_bounds__(32) void params_gemm_wmma(
    const float* __restrict__ hs, const float* __restrict__ Wc,
    const float* __restrict__ bctrl, float* __restrict__ params) {
  const int lane = threadIdx.x & 31;
  const int row0 = blockIdx.x * 16;          // 50 tiles of M
  const int col0 = blockIdx.y * 16;          // 11 tiles of N (169 -> pad 176)
  const int mn   = lane & 15;
  const bool lo  = lane < 16;
  const int aoff = lo ? 0 : 8;               // unused: A also loads via koff, low half K0..15, high K16..31
  const int koff = lo ? 0 : 16;              // B: low half K0..15, high K16..31
  const int col  = col0 + mn;
  const int colc = (col > NP - 1) ? (NP - 1) : col;  // clamp to stay in-bounds

  v8f acc = fzero8();
  const float* arow = hs + (size_t)(row0 + mn) * HID;
  const float* brow = Wc + (size_t)colc * HID;
#pragma unroll
  for (int kt = 0; kt < HID / 32; ++kt) {
    const int k0 = kt * 32;
    v16h a, b;
    const float* ap = arow + k0 + koff;
#pragma unroll
    for (int j = 0; j < 8; ++j) {
      a[j]     = (_Float16)ap[j];
      a[8 + j] = (_Float16)ap[8 + j];
    }
    const float* bp = brow + k0 + koff;
#pragma unroll
    for (int j = 0; j < 16; ++j) b[j] = (_Float16)bp[j];
    if (col > NP - 1) b = hzero16();         // zero padded columns
    acc = __builtin_amdgcn_wmma_f32_16x16x32_f16(false, a, false, b,
                                                 (short)0, acc, false, false);
  }
  if (col <= NP - 1) {
    const int rb = lo ? 0 : 8;               // D: lanes 0-15 rows 0-7, 16-31 rows 8-15
    const float bias = bctrl[col];
#pragma unroll
    for (int r = 0; r < 8; ++r)
      params[(size_t)(row0 + rb + r) * NP + col] = acc[r] + bias;
  }
}

// ---------------------------------------------------------------------------
// Kernel 2: dynamic 3-layer 1x1-conv MLP, weights-in-A / pixels-in-B mapping.
//   D0 = w0(A, rows=out ch) x x(B, cols=pixels)  -> lane n holds pixel n's
//   channels in VGPRs 0..7, which IS the next layer's B layout: no transpose.
// Two 16-pixel tiles per iteration so all 32 lanes load coalesced 128B rows;
// shfl_xor(16) moves the upper half-wave's tile into B-lanes 0-15.
// ---------------------------------------------------------------------------
__global__ __launch_bounds__(256) void seg_head_wmma(
    const float* __restrict__ mask, const float* __restrict__ refs,
    const float* __restrict__ sizes, const float* __restrict__ params,
    float* __restrict__ out) {
  __shared__ float sp[NP];

  const int inst = blockIdx.x;
  const int bt   = inst / QQN;
  const int qq   = inst - bt * QQN;
  const int bb   = bt / TTN;
  const int lane = threadIdx.x & 31;
  const int wv   = threadIdx.x >> 5;
  const int m    = lane & 15;
  const bool lo  = lane < 16;

  for (int i = threadIdx.x; i < NP; i += blockDim.x)
    sp[i] = params[(size_t)inst * NP + i];
  __syncthreads();

  // A operands = per-instance weights (rows 8..15 zero).
  // A layout: lane m<16 holds K0..15; lane m+16 holds K16..31.
  v16h A0 = hzero16(), A1 = hzero16();
  if (m < 8) {
    if (lo) {
#pragma unroll
      for (int k = 0; k < 8; ++k) A0[k] = (_Float16)sp[m * 10 + k];      // w0[m][0..7]
#pragma unroll
      for (int k = 0; k < 8; ++k) A1[k] = (_Float16)sp[80 + m * 8 + k];  // w1[m][0..7]
      A0[8] = (_Float16)sp[m * 10 + 8];                                  // w0[m][8]
      A0[9] = (_Float16)sp[m * 10 + 9];                                  // w0[m][9]
    }
  }
  // Wave-uniform per-row constants.
  float b0r[8], b1r[8], w2r[8];
#pragma unroll
  for (int r = 0; r < 8; ++r) {
    b0r[r] = sp[152 + r];
    b1r[r] = sp[160 + r];
    w2r[r] = sp[144 + r];
  }
  const float b2   = sp[168];
  const float refx = refs[((size_t)bt * QQN + qq) * 2 + 0] * sizes[bb * 2 + 1]; // * W*4
  const float refy = refs[((size_t)bt * QQN + qq) * 2 + 1] * sizes[bb * 2 + 0]; // * H*4
  const float* mbase = mask + (size_t)bt * 8 * HWN;
  float* obase = out + (size_t)inst * HWN;

  for (int tp = blockIdx.y * 8 + wv; tp < NPAIRS; tp += gridDim.y * 8) {
    const int p0  = tp * 32;
    const int pix = p0 + lane;               // every lane owns one pixel

    // Coalesced feature loads: 8 x 128B rows.
    const float* mp = mbase + pix;
    float c[8];
#pragma unroll
    for (int ch = 0; ch < 8; ++ch) c[ch] = mp[ch * HWN];
    const int hy = pix / WWN;
    const int wx = pix - hy * WWN;
    const float rx = refx - (float)(wx * 4 + 2);
    const float ry = refy - (float)(hy * 4 + 2);

    // Pack this pixel's K=10 column (f16): halves 0..7 = ch0..7, 8..9 = rel.
    HU own;
    own.h = hzero16();
#pragma unroll
    for (int ch = 0; ch < 8; ++ch) own.h[ch] = (_Float16)c[ch];
    own.h[8] = (_Float16)rx;
    own.h[9] = (_Float16)ry;

    // Swap halves so lanes 0-15 can also see the upper tile's columns.
    unsigned int sw[5];
#pragma unroll
    for (int j = 0; j < 5; ++j)
      sw[j] = (unsigned int)__shfl_xor((int)own.u[j], 16, 32);

    HU Bt0, Bt1;
#pragma unroll
    for (int j = 0; j < 8; ++j) {
      Bt0.u[j] = lo ? own.u[j] : 0u;                  // K16..31 lanes -> zero
      Bt1.u[j] = (lo && j < 5) ? sw[j] : 0u;
    }

    // Layer 0 (K=10 -> 8 channels), both tiles.
    v8f d0a = __builtin_amdgcn_wmma_f32_16x16x32_f16(false, A0, false, Bt0.h,
                                                     (short)0, fzero8(), false, false);
    v8f d0b = __builtin_amdgcn_wmma_f32_16x16x32_f16(false, A0, false, Bt1.h,
                                                     (short)0, fzero8(), false, false);

    // relu+bias, repack as next-layer B (already channel-major per lane).
    HU h1a, h1b;
    h1a.h = hzero16();
    h1b.h = hzero16();
#pragma unroll
    for (int r = 0; r < 8; ++r) {
      float va = d0a[r] + b0r[r]; va = va > 0.f ? va : 0.f;
      float vb = d0b[r] + b0r[r]; vb = vb > 0.f ? vb : 0.f;
      h1a.h[r] = (_Float16)va;
      h1b.h[r] = (_Float16)vb;
    }
#pragma unroll
    for (int j = 0; j < 4; ++j) {            // zero the dead upper half-wave
      h1a.u[j] = lo ? h1a.u[j] : 0u;
      h1b.u[j] = lo ? h1b.u[j] : 0u;
    }

    // Layer 1 (8 -> 8), both tiles.
    v8f d1a = __builtin_amdgcn_wmma_f32_16x16x32_f16(false, A1, false, h1a.h,
                                                     (short)0, fzero8(), false, false);
    v8f d1b = __builtin_amdgcn_wmma_f32_16x16x32_f16(false, A1, false, h1b.h,
                                                     (short)0, fzero8(), false, false);

    // Layer 2 (8 -> 1): purely per-lane dot over accumulator VGPRs.
    float o0 = b2, o1 = b2;
#pragma unroll
    for (int r = 0; r < 8; ++r) {
      float va = d1a[r] + b1r[r]; va = va > 0.f ? va : 0.f;
      float vb = d1b[r] + b1r[r]; vb = vb > 0.f ? vb : 0.f;
      o0 = fmaf(va, w2r[r], o0);
      o1 = fmaf(vb, w2r[r], o1);
    }
    // Move tile1 results to lanes 16-31, store one contiguous 128B row.
    const float o1sw = __shfl_xor(o1, 16, 32);
    obase[pix] = lo ? o0 : o1sw;
  }
}

extern "C" void kernel_launch(void* const* d_in, const int* in_sizes, int n_in,
                              void* d_out, int out_size, void* d_ws, size_t ws_size,
                              hipStream_t stream) {
  const float* hs    = (const float*)d_in[0];   // (16,50,256)
  const float* mask  = (const float*)d_in[1];   // (16,8,72,120)
  const float* refs  = (const float*)d_in[2];   // (16,50,2)
  const float* sizes = (const float*)d_in[3];   // (2,2)
  const float* Wc    = (const float*)d_in[4];   // (169,256)
  const float* bctrl = (const float*)d_in[5];   // (169,)
  float* out = (float*)d_out;                   // (16,50,72,120)
  float* pw  = (float*)d_ws;                    // params scratch: 800*169 f32

  params_gemm_wmma<<<dim3(NINST / 16, (NP + 15) / 16), 32, 0, stream>>>(
      hs, Wc, bctrl, pw);
  seg_head_wmma<<<dim3(NINST, 4), 256, 0, stream>>>(mask, refs, sizes, pw, out);
}
